// MultiHeadAttn_17678085390590
// MI455X (gfx1250) — compile-verified, matrix operand order checked by reading, not run on hardware
//
#include <hip/hip_runtime.h>
#include <hip/hip_bf16.h>

// MI455X (gfx1250) fused multi-head attention forward.
// B=4, L=2048, HID=1024, NHEADS=16, HEAD_DIM=64.
// All GEMMs on v_wmma_f32_16x16x32_f16 (f16 in, f32 accumulate).
// Causal mask computed analytically (reference mask is tril).
//
// gfx1250-specific data movement:
//  - TDM tensor_load_to_lds (6-arg form) for strided 2D row-major tile
//    copies (Q tile, out-proj A tile), D# packed per ISA 8.3/8.4.
//  - global_load_async_to_lds_b128 for the K tile frag-ready scatter
//    (per-lane LDS destinations), ASYNCcnt synchronized.
//  - global_prefetch_b8 via __builtin_prefetch for next tiles.

typedef __attribute__((ext_vector_type(16))) _Float16 v16h;
typedef __attribute__((ext_vector_type(2)))  __fp16   v2fp16;
typedef __attribute__((ext_vector_type(8)))  float    v8f;
typedef __attribute__((ext_vector_type(4)))  unsigned int v4u;
typedef __attribute__((ext_vector_type(8)))  int      v8i_;
typedef __attribute__((ext_vector_type(4)))  int      v4i_;

typedef __attribute__((address_space(1))) v4i_ as1_v4i;
typedef __attribute__((address_space(3))) v4i_ as3_v4i;

#define BATCH    4
#define SEQ      2048
#define HID      1024
#define NHEADS   16
#define HDIM     64
#define MROWS    (BATCH * SEQ)      // 8192
#define SCALE    0.125f             // 1/sqrt(64)
#define NEG_INF  (-__builtin_inff())

#if __has_builtin(__builtin_amdgcn_global_load_async_to_lds_b128)
#define HAVE_ASYNC_LDS 1
#else
#define HAVE_ASYNC_LDS 0
#endif
#if __has_builtin(__builtin_amdgcn_tensor_load_to_lds)
#define HAVE_TDM 1
#else
#define HAVE_TDM 0
#endif
#if __has_builtin(__builtin_amdgcn_s_wait_asynccnt)
#define HAVE_WAIT_ASYNC 1
#else
#define HAVE_WAIT_ASYNC 0
#endif
#if __has_builtin(__builtin_amdgcn_s_wait_tensorcnt)
#define HAVE_WAIT_TENSOR 1
#else
#define HAVE_WAIT_TENSOR 0
#endif
#if __has_builtin(__builtin_amdgcn_cvt_pkrtz)
#define HAVE_PKRTZ 1
#else
#define HAVE_PKRTZ 0
#endif

__device__ __forceinline__ void wait_async0() {
#if HAVE_WAIT_ASYNC
  __builtin_amdgcn_s_wait_asynccnt(0);
#else
  asm volatile("s_wait_asynccnt 0x0" ::: "memory");
#endif
}
__device__ __forceinline__ void wait_tensor0() {
#if HAVE_WAIT_TENSOR
  __builtin_amdgcn_s_wait_tensorcnt(0);
#else
  asm volatile("s_wait_tensorcnt 0x0" ::: "memory");
#endif
}

#if HAVE_ASYNC_LDS
__device__ __forceinline__ void async_copy_b128(const void* g, void* l) {
  __builtin_amdgcn_global_load_async_to_lds_b128((as1_v4i*)g, (as3_v4i*)l, 0,
                                                 0);
}
#endif

#if HAVE_TDM
// TDM 2D f16 tile load: tile_h rows x tile_w elems, row stride in elems.
// D# per CDNA5 ISA 8.3/8.4: group0 {count=1, lds_addr, global_addr, type=2},
// group1 {data_size=2B, tensor dims, tile dims, dim0 stride}. 2D tensor ->
// groups 2/3 (and trailing v8i arg) zero. Issue from one wave; EXEC ignored.
__device__ __forceinline__ void tdm_load_2d_f16(const _Float16* gsrc,
                                                _Float16* lds_dst,
                                                unsigned tile_w,
                                                unsigned tile_h,
                                                unsigned row_stride) {
  const unsigned long long ga = (unsigned long long)(uintptr_t)gsrc;
  const unsigned lds_addr =
      (unsigned)(uintptr_t)(__attribute__((address_space(3))) _Float16*)lds_dst;
  v4u g0;
  g0[0] = 1u;                                   // count=1, user descriptor
  g0[1] = lds_addr;                             // bits 63:32
  g0[2] = (unsigned)ga;                         // global_addr[31:0]
  g0[3] = (unsigned)(ga >> 32) | (2u << 30);    // addr[56:32] | type=2
  v8i_ g1;
  g1[0] = 1 << 16;                              // data_size=1 -> 2 bytes
  g1[1] = (int)((tile_w & 0xFFFFu) << 16);      // tensor_dim0 lo16 @63:48
  g1[2] = (int)((tile_w >> 16) | ((tile_h & 0xFFFFu) << 16));  // d0 hi|d1 lo
  g1[3] = (int)((tile_h >> 16) | (tile_w << 16));  // d1 hi | tile_dim0
  g1[4] = (int)(tile_h & 0xFFFFu);              // tile_dim1 (tile_dim2=0)
  g1[5] = (int)row_stride;                      // tensor_dim0_stride lo32
  g1[6] = 0;
  g1[7] = 0;
  const v4i_ gz4 = {0, 0, 0, 0};
  const v8i_ gz8 = {0, 0, 0, 0, 0, 0, 0, 0};
  __builtin_amdgcn_tensor_load_to_lds(g0, g1, gz4, gz4, gz8, 0);
}
#endif

__device__ __forceinline__ v8f wmma16x16x32(v16h a, v16h b, v8f c) {
  return __builtin_amdgcn_wmma_f32_16x16x32_f16(
      /*neg_a=*/false, a, /*neg_b=*/false, b,
      /*c_mod=*/(short)0, c, /*reuse_a=*/false, /*reuse_b=*/false);
}

// A fragment (16x32 f16) from row-major LDS -> 2x ds_load_b128 per lane.
__device__ __forceinline__ v16h load_a_frag(const _Float16* base, int stride,
                                            int lane) {
  const int row = lane & 15;
  const int kb  = (lane & 16) ? 8 : 0;
  v16h a;
#pragma unroll
  for (int v = 0; v < 8; ++v) {
    const int k = kb + ((v < 4) ? (2 * v) : (16 + 2 * (v - 4)));
    a[2 * v]     = base[row * stride + k];
    a[2 * v + 1] = base[row * stride + k + 1];
  }
  return a;
}

// B fragment from frag-ready LDS region: 32B contiguous read per lane.
__device__ __forceinline__ v16h load_b_ready(const _Float16* base, int lane) {
  return *(const v16h*)(base + lane * 16);
}

// Pack 8 f32 into 8 f16 (one uint4) using v_cvt_pk_f16_f32 when available.
__device__ __forceinline__ uint4 pack8h(float4 a, float4 b) {
  union { v2fp16 h2[4]; _Float16 h[8]; uint4 u; } u;
#if HAVE_PKRTZ
  u.h2[0] = __builtin_amdgcn_cvt_pkrtz(a.x, a.y);
  u.h2[1] = __builtin_amdgcn_cvt_pkrtz(a.z, a.w);
  u.h2[2] = __builtin_amdgcn_cvt_pkrtz(b.x, b.y);
  u.h2[3] = __builtin_amdgcn_cvt_pkrtz(b.z, b.w);
#else
  u.h[0] = (_Float16)a.x; u.h[1] = (_Float16)a.y;
  u.h[2] = (_Float16)a.z; u.h[3] = (_Float16)a.w;
  u.h[4] = (_Float16)b.x; u.h[5] = (_Float16)b.y;
  u.h[6] = (_Float16)b.z; u.h[7] = (_Float16)b.w;
#endif
  return u.u;
}

// ---------------------------------------------------------------------------
// Kernel 1: fused QKV projection.  Y = X @ W^T + bias, cast to f16.
// Grid: (HID/64, MROWS/128, 3)  Block: 128 threads = 4 waves.
// ---------------------------------------------------------------------------
__global__ __launch_bounds__(128) void qkv_proj_kernel(
    const float* __restrict__ xq, const float* __restrict__ xk,
    const float* __restrict__ xv,
    const float* __restrict__ Wq, const float* __restrict__ Wk,
    const float* __restrict__ Wv,
    const float* __restrict__ bq, const float* __restrict__ bk,
    const float* __restrict__ bv,
    _Float16* __restrict__ dq, _Float16* __restrict__ dk,
    _Float16* __restrict__ dv) {
  const int z = blockIdx.z;
  const float* X    = (z == 0) ? xq : ((z == 1) ? xk : xv);
  const float* W    = (z == 0) ? Wq : ((z == 1) ? Wk : Wv);
  const float* bias = (z == 0) ? bq : ((z == 1) ? bk : bv);
  _Float16*    D    = (z == 0) ? dq : ((z == 1) ? dk : dv);

  const int n0   = blockIdx.x * 64;
  const int m0   = blockIdx.y * 128;
  const int t    = threadIdx.x;
  const int lane = t & 31;
  const int w    = t >> 5;

  __shared__ alignas(32) _Float16 Asub[128 * 32];      // [m][k] row-major
  __shared__ alignas(32) _Float16 Bfrag[4][32][16];    // frag-ready W^T tiles

  v8f acc[2][4];
  const v8f zero = {};
#pragma unroll
  for (int s = 0; s < 2; ++s)
#pragma unroll
    for (int i = 0; i < 4; ++i) acc[s][i] = zero;

  const float* asrc = X + (size_t)(m0 + t) * HID;
  const int    bn   = t >> 1;
  const int    bkb  = (t & 1) * 16;
  const float* bsrc = W + (size_t)(n0 + bn) * HID + bkb;

  for (int k0 = 0; k0 < HID; k0 += 32) {
    __syncthreads();
    {  // A tile: 128x32 f32 -> f16 (v_cvt_pk_f16_f32), b128 ds stores
      _Float16* dst = &Asub[t * 32];
#pragma unroll
      for (int i = 0; i < 4; ++i) {
        const float4 fa = ((const float4*)(asrc + k0))[2 * i];
        const float4 fb = ((const float4*)(asrc + k0))[2 * i + 1];
        *(uint4*)&dst[8 * i] = pack8h(fa, fb);
      }
    }
    {  // B tile -> frag-ready layout
      const float4 f0 = ((const float4*)(bsrc + k0))[0];
      const float4 f1 = ((const float4*)(bsrc + k0))[1];
      const float4 f2 = ((const float4*)(bsrc + k0))[2];
      const float4 f3 = ((const float4*)(bsrc + k0))[3];
      const int nt = bn >> 4, col = bn & 15, eb = bkb ? 8 : 0;
      *(uint4*)&Bfrag[nt][col][eb]      = pack8h(f0, f1);
      *(uint4*)&Bfrag[nt][col + 16][eb] = pack8h(f2, f3);
    }
    if (k0 + 32 < HID) {  // global_prefetch_b8 of next K-step
      __builtin_prefetch(asrc + k0 + 32, 0, 1);
      __builtin_prefetch(bsrc + k0 + 32, 0, 1);
    }
    __syncthreads();

    const v16h a0 = load_a_frag(&Asub[(w * 32) * 32], 32, lane);
    const v16h a1 = load_a_frag(&Asub[(w * 32 + 16) * 32], 32, lane);
#pragma unroll
    for (int nt = 0; nt < 4; ++nt) {
      const v16h bf = load_b_ready(&Bfrag[nt][0][0], lane);
      acc[0][nt] = wmma16x16x32(a0, bf, acc[0][nt]);
      acc[1][nt] = wmma16x16x32(a1, bf, acc[1][nt]);
    }
  }

  const int colb   = lane & 15;
  const int rowoff = (lane & 16) ? 8 : 0;
#pragma unroll
  for (int nt = 0; nt < 4; ++nt) {
    const int n  = n0 + nt * 16 + colb;
    const float bb = bias[n];
#pragma unroll
    for (int s = 0; s < 2; ++s)
#pragma unroll
      for (int e = 0; e < 8; ++e) {
        const int row = m0 + w * 32 + s * 16 + rowoff + e;
        D[(size_t)row * HID + n] = (_Float16)(acc[s][nt][e] + bb);
      }
  }
}

// ---------------------------------------------------------------------------
// Kernel 2: causal flash attention per (b,h).
// Grid: (SEQ/64, BATCH*NHEADS)  Block: 128 threads = 4 waves.
// ---------------------------------------------------------------------------
__global__ __launch_bounds__(128) void attn_kernel(
    const _Float16* __restrict__ Q, const _Float16* __restrict__ K,
    const _Float16* __restrict__ V, _Float16* __restrict__ O) {
  const int q0   = blockIdx.x * 64;
  const int bh   = blockIdx.y;
  const int b    = bh >> 4;
  const int h    = bh & 15;
  const int t    = threadIdx.x;
  const int lane = t & 31;
  const int w    = t >> 5;

  __shared__ alignas(32) _Float16 Qs[64 * 64];          // [qrow][d] row-major
  __shared__ alignas(32) _Float16 Kfrag[2][2][32][16];  // frag-ready K^T
  __shared__ alignas(32) _Float16 Vfrag[4][32][16];     // frag-ready V
  __shared__ alignas(32) _Float16 Ps[4][16 * 32];       // per-wave P (A-layout)

  const size_t headbase = ((size_t)b * SEQ * NHEADS + h) * HDIM;  // + l*HID + d

#if HAVE_TDM
  if (w == 0) {  // TDM: 64x64 f16 tile, row stride HID, into Qs
    tdm_load_2d_f16(Q + headbase + (size_t)q0 * HID, Qs, 64, 64, HID);
    wait_tensor0();
  }
#else
  {  // Q tile 64x64 straight copy
    const int r  = t >> 1;
    const int cb = (t & 1) * 32;
    const uint4* src =
        (const uint4*)(Q + headbase + (size_t)(q0 + r) * HID + cb);
    uint4* dst = (uint4*)&Qs[r * 64 + cb];
    dst[0] = src[0]; dst[1] = src[1]; dst[2] = src[2]; dst[3] = src[3];
  }
#endif
  __syncthreads();

  const v16h aq0 = load_a_frag(&Qs[w * 16 * 64], 64, lane);       // d 0..31
  const v16h aq1 = load_a_frag(&Qs[w * 16 * 64 + 32], 64, lane);  // d 32..63

  v8f o0 = {}, o1 = {}, o2 = {}, o3 = {};
  float m_[8], l_[8];
#pragma unroll
  for (int e = 0; e < 8; ++e) { m_[e] = NEG_INF; l_[e] = 0.0f; }

  const int colb   = lane & 15;
  const int rowoff = (lane & 16) ? 8 : 0;
  const int nblk   = (q0 >> 5) + 2;  // causal: keys up to q0+63

  const int ldkey = t >> 2;          // 0..31
  const int lddb  = (t & 3) * 16;    // 0,16,32,48

  for (int blk = 0; blk < nblk; ++blk) {
    const int j0 = blk * 32;
    __syncthreads();
    {  // K tile -> frag-ready: per-lane scatter, async-to-LDS when available
      const _Float16* gsrc = K + headbase + (size_t)(j0 + ldkey) * HID + lddb;
      const int c = ldkey >> 4, col = ldkey & 15;
      const int slab = lddb >> 5;
      const int eb = (lddb & 16) ? 8 : 0;
#if HAVE_ASYNC_LDS
      async_copy_b128(gsrc,     &Kfrag[c][slab][col][eb]);
      async_copy_b128(gsrc + 8, &Kfrag[c][slab][col + 16][eb]);
#else
      const uint4* src = (const uint4*)gsrc;
      *(uint4*)&Kfrag[c][slab][col][eb]      = src[0];
      *(uint4*)&Kfrag[c][slab][col + 16][eb] = src[1];
#endif
    }
    {  // V tile -> frag-ready: element scatter (2B granularity, manual)
      const uint4* src =
          (const uint4*)(V + headbase + (size_t)(j0 + ldkey) * HID + lddb);
      _Float16 tmp[16];
      *((uint4*)tmp)       = src[0];
      *(((uint4*)tmp) + 1) = src[1];
      const int half = (ldkey >> 3) & 1;
      const int e    = (ldkey & 7) + ((ldkey & 16) ? 8 : 0);
      const int nt   = lddb >> 4;
#pragma unroll
      for (int i = 0; i < 16; ++i) Vfrag[nt][i + 16 * half][e] = tmp[i];
    }
    if (blk + 1 < nblk) {  // global_prefetch_b8 of next key block
      __builtin_prefetch(
          K + headbase + (size_t)(j0 + 32 + ldkey) * HID + lddb, 0, 1);
      __builtin_prefetch(
          V + headbase + (size_t)(j0 + 32 + ldkey) * HID + lddb, 0, 1);
    }
#if HAVE_ASYNC_LDS
    wait_async0();
#endif
    __syncthreads();

    // S(16x32) = Q_wave(16x64) @ K_blk^T
    v8f s0 = {}, s1 = {};
    {
      const v16h b00 = load_b_ready(&Kfrag[0][0][0][0], lane);
      const v16h b01 = load_b_ready(&Kfrag[0][1][0][0], lane);
      const v16h b10 = load_b_ready(&Kfrag[1][0][0][0], lane);
      const v16h b11 = load_b_ready(&Kfrag[1][1][0][0], lane);
      s0 = wmma16x16x32(aq0, b00, s0);
      s0 = wmma16x16x32(aq1, b01, s0);
      s1 = wmma16x16x32(aq0, b10, s1);
      s1 = wmma16x16x32(aq1, b11, s1);
    }

    // Online softmax (rows = element idx + lane-half; 16 cols across lanes).
#pragma unroll
    for (int e = 0; e < 8; ++e) {
      const int row = q0 + w * 16 + rowoff + e;
      const int c0  = j0 + colb;
      const int c1  = j0 + 16 + colb;
      float v0 = (c0 <= row) ? s0[e] * SCALE : NEG_INF;
      float v1 = (c1 <= row) ? s1[e] * SCALE : NEG_INF;

      float mx = fmaxf(v0, v1);
#pragma unroll
      for (int d = 1; d < 16; d <<= 1) mx = fmaxf(mx, __shfl_xor(mx, d, 32));

      const float mnew  = fmaxf(m_[e], mx);
      const float alpha = __expf(m_[e] - mnew);
      const float p0    = __expf(v0 - mnew);
      const float p1    = __expf(v1 - mnew);

      float rs = p0 + p1;
#pragma unroll
      for (int d = 1; d < 16; d <<= 1) rs += __shfl_xor(rs, d, 32);

      l_[e] = l_[e] * alpha + rs;
      m_[e] = mnew;
      o0[e] *= alpha; o1[e] *= alpha; o2[e] *= alpha; o3[e] *= alpha;

      Ps[w][(rowoff + e) * 32 + colb]      = (_Float16)p0;
      Ps[w][(rowoff + e) * 32 + 16 + colb] = (_Float16)p1;
    }

    // O(16x64) += P(16x32) @ V(32x64)
    const v16h ap = load_a_frag(Ps[w], 32, lane);
    {
      const v16h bv0 = load_b_ready(&Vfrag[0][0][0], lane);
      const v16h bv1 = load_b_ready(&Vfrag[1][0][0], lane);
      const v16h bv2 = load_b_ready(&Vfrag[2][0][0], lane);
      const v16h bv3 = load_b_ready(&Vfrag[3][0][0], lane);
      o0 = wmma16x16x32(ap, bv0, o0);
      o1 = wmma16x16x32(ap, bv1, o1);
      o2 = wmma16x16x32(ap, bv2, o2);
      o3 = wmma16x16x32(ap, bv3, o3);
    }
  }

  float linv[8];
#pragma unroll
  for (int e = 0; e < 8; ++e) linv[e] = 1.0f / l_[e];
#pragma unroll
  for (int e = 0; e < 8; ++e) {
    const int row = q0 + w * 16 + rowoff + e;
    _Float16* dst = O + headbase + (size_t)row * HID;
    dst[0 * 16 + colb] = (_Float16)(o0[e] * linv[e]);
    dst[1 * 16 + colb] = (_Float16)(o1[e] * linv[e]);
    dst[2 * 16 + colb] = (_Float16)(o2[e] * linv[e]);
    dst[3 * 16 + colb] = (_Float16)(o3[e] * linv[e]);
  }
}

// ---------------------------------------------------------------------------
// Kernel 3: output projection.  out = attn(f16) @ Wo^T + bo, f32 result.
// Grid: (HID/64, MROWS/128)  Block: 128 threads = 4 waves.
// ---------------------------------------------------------------------------
__global__ __launch_bounds__(128) void out_proj_kernel(
    const _Float16* __restrict__ A, const float* __restrict__ Wo,
    const float* __restrict__ bo, float* __restrict__ out) {
  const int n0   = blockIdx.x * 64;
  const int m0   = blockIdx.y * 128;
  const int t    = threadIdx.x;
  const int lane = t & 31;
  const int w    = t >> 5;

  __shared__ alignas(32) _Float16 Asub[128 * 32];
  __shared__ alignas(32) _Float16 Bfrag[4][32][16];

  v8f acc[2][4];
  const v8f zero = {};
#pragma unroll
  for (int s = 0; s < 2; ++s)
#pragma unroll
    for (int i = 0; i < 4; ++i) acc[s][i] = zero;

  const _Float16* asrc = A + (size_t)(m0 + t) * HID;
  const int    bn   = t >> 1;
  const int    bkb  = (t & 1) * 16;
  const float* bsrc = Wo + (size_t)(n0 + bn) * HID + bkb;

  for (int k0 = 0; k0 < HID; k0 += 32) {
    __syncthreads();
#if HAVE_TDM
    if (w == 0) {  // TDM: 128 rows x 32 f16, row stride HID, into Asub
      tdm_load_2d_f16(A + (size_t)m0 * HID + k0, Asub, 32, 128, HID);
      wait_tensor0();
    }
#else
    {  // A tile straight f16 copy
      const uint4* src = (const uint4*)(asrc + k0);
      uint4* dst = (uint4*)&Asub[t * 32];
      dst[0] = src[0]; dst[1] = src[1]; dst[2] = src[2]; dst[3] = src[3];
    }
#endif
    {  // B tile -> frag-ready
      const float4 f0 = ((const float4*)(bsrc + k0))[0];
      const float4 f1 = ((const float4*)(bsrc + k0))[1];
      const float4 f2 = ((const float4*)(bsrc + k0))[2];
      const float4 f3 = ((const float4*)(bsrc + k0))[3];
      const int nt = bn >> 4, col = bn & 15, eb = bkb ? 8 : 0;
      *(uint4*)&Bfrag[nt][col][eb]      = pack8h(f0, f1);
      *(uint4*)&Bfrag[nt][col + 16][eb] = pack8h(f2, f3);
    }
    if (k0 + 32 < HID) {
      __builtin_prefetch(asrc + k0 + 32, 0, 1);
      __builtin_prefetch(bsrc + k0 + 32, 0, 1);
    }
    __syncthreads();

    const v16h a0 = load_a_frag(&Asub[(w * 32) * 32], 32, lane);
    const v16h a1 = load_a_frag(&Asub[(w * 32 + 16) * 32], 32, lane);
#pragma unroll
    for (int nt = 0; nt < 4; ++nt) {
      const v16h bf = load_b_ready(&Bfrag[nt][0][0], lane);
      acc[0][nt] = wmma16x16x32(a0, bf, acc[0][nt]);
      acc[1][nt] = wmma16x16x32(a1, bf, acc[1][nt]);
    }
  }

  const int colb   = lane & 15;
  const int rowoff = (lane & 16) ? 8 : 0;
#pragma unroll
  for (int nt = 0; nt < 4; ++nt) {
    const int n  = n0 + nt * 16 + colb;
    const float bb = bo[n];
#pragma unroll
    for (int s = 0; s < 2; ++s)
#pragma unroll
      for (int e = 0; e < 8; ++e) {
        const int row = m0 + w * 32 + s * 16 + rowoff + e;
        out[(size_t)row * HID + n] = acc[s][nt][e] + bb;
      }
  }
}

// ---------------------------------------------------------------------------
extern "C" void kernel_launch(void* const* d_in, const int* in_sizes, int n_in,
                              void* d_out, int out_size, void* d_ws,
                              size_t ws_size, hipStream_t stream) {
  const float* query = (const float*)d_in[0];
  const float* key   = (const float*)d_in[1];
  const float* val   = (const float*)d_in[2];
  // d_in[3] = attn_mask (tril) -> handled analytically in attn_kernel
  const float* Wq = (const float*)d_in[4];
  const float* bq = (const float*)d_in[5];
  const float* Wk = (const float*)d_in[6];
  const float* bk = (const float*)d_in[7];
  const float* Wv = (const float*)d_in[8];
  const float* bv = (const float*)d_in[9];
  const float* Wo = (const float*)d_in[10];
  const float* bo = (const float*)d_in[11];

  const size_t elems = (size_t)MROWS * HID;  // 8M f16 per buffer
  _Float16* qf = (_Float16*)d_ws;
  _Float16* kf = qf + elems;
  _Float16* vf = kf + elems;
  _Float16* af = vf + elems;

  qkv_proj_kernel<<<dim3(HID / 64, MROWS / 128, 3), 128, 0, stream>>>(
      query, key, val, Wq, Wk, Wv, bq, bk, bv, qf, kf, vf);

  attn_kernel<<<dim3(SEQ / 64, BATCH * NHEADS), 128, 0, stream>>>(qf, kf, vf,
                                                                  af);

  out_proj_kernel<<<dim3(HID / 64, MROWS / 128), 128, 0, stream>>>(
      af, Wo, bo, (float*)d_out);
}
